// Attention_89043261981203
// MI455X (gfx1250) — matrix operand order repaired and checked by reading, not compiled, not run on hardware
//
#include <hip/hip_runtime.h>
#include <hip/hip_bf16.h>
#include <cstdint>

typedef __bf16 bf16;
typedef __attribute__((ext_vector_type(16))) __bf16 v16bf;
typedef __attribute__((ext_vector_type(8)))  __bf16 v8bf;
typedef __attribute__((ext_vector_type(4)))  __bf16 v4bf;
typedef __attribute__((ext_vector_type(8)))  float  v8f;
typedef __attribute__((ext_vector_type(4)))  float  v4f;
typedef __attribute__((ext_vector_type(4)))  int    v4i;

// Problem dims (fixed by the reference)
#define D_MODEL 1024
#define NSEQ    2048
#define BATCH   2
#define NHEADS  16
#define DHEAD   64
#define MROWS   (BATCH*NSEQ)   // 4096 rows for the big GEMMs

// ---------------------------------------------------------------------------
// CDNA5 async global->LDS path (ASYNCcnt-tracked), with sync fallback.
// Builtin prototype (from hipcc diagnostic): b128 variant takes
//   (int4 addrspace(1)* gsrc, int4 addrspace(3)* lds, imm int, imm int)
// ---------------------------------------------------------------------------
#if defined(__has_builtin)
#  if __has_builtin(__builtin_amdgcn_global_load_async_to_lds_b128)
#    define HAVE_ASYNC_LDS 1
#  endif
#endif
#ifndef HAVE_ASYNC_LDS
#  define HAVE_ASYNC_LDS 0
#endif

typedef __attribute__((address_space(1))) v4i* g_v4i_ptr;   // pointer to global int4
typedef __attribute__((address_space(3))) v4i* l_v4i_ptr;   // pointer to LDS int4

// copy 16 bytes global -> LDS (per lane)
__device__ __forceinline__ void copy_b128_to_lds(void* ldsDst, const void* gSrc) {
#if HAVE_ASYNC_LDS
  __builtin_amdgcn_global_load_async_to_lds_b128((g_v4i_ptr)(void*)gSrc,
                                                 (l_v4i_ptr)ldsDst, 0, 0);
#else
  *(v8bf*)ldsDst = *(const v8bf*)gSrc;
#endif
}

__device__ __forceinline__ void wait_async0() {
#if HAVE_ASYNC_LDS
#  if __has_builtin(__builtin_amdgcn_s_wait_asynccnt)
  __builtin_amdgcn_s_wait_asynccnt(0);
#  else
  asm volatile("s_wait_asynccnt 0x0" ::: "memory");
#  endif
#endif
}

// ---------------------------------------------------------------------------
// WMMA helpers: D = A(16x32 bf16) x B(32x16 bf16) + C(16x16 f32), wave32.
// ---------------------------------------------------------------------------
__device__ __forceinline__ v8f wmma_bf16(v16bf a, v16bf b, v8f c) {
  return __builtin_amdgcn_wmma_f32_16x16x32_bf16(
      /*neg_a=*/false, a, /*neg_b=*/false, b,
      /*c_mod=*/(short)0, c, /*reuse_a=*/false, /*reuse_b=*/false);
}

// A fragment (16x32, 16-bit): lane half h holds K = h*8..h*8+7 and 16+h*8..23+h*8
__device__ __forceinline__ v16bf frag_a(const bf16* base, int half) {
  union { v16bf v; v8bf h[2]; } u;
  u.h[0] = *(const v8bf*)(base + half * 8);
  u.h[1] = *(const v8bf*)(base + 16 + half * 8);
  return u.v;
}
// B fragment (32x16, 16-bit): lane half h holds K = h*8..h*8+7 and 16+h*8..23+h*8
__device__ __forceinline__ v16bf frag_b(const bf16* base, int half) {
  union { v16bf v; v8bf h[2]; } u;
  u.h[0] = *(const v8bf*)(base + half * 8);
  u.h[1] = *(const v8bf*)(base + 16 + half * 8);
  return u.v;
}

// ---------------------------------------------------------------------------
// f32 -> bf16 conversion (vector x4 per thread)
// ---------------------------------------------------------------------------
__global__ void cvt_f32_bf16(const float* __restrict__ src,
                             unsigned short* __restrict__ dst_, int n4) {
  bf16* dst = (bf16*)dst_;
  int i = blockIdx.x * blockDim.x + threadIdx.x;
  if (i < n4) {
    v4f f = *(const v4f*)(src + (size_t)i * 4);
    v4bf o;
    o.x = (bf16)f.x; o.y = (bf16)f.y; o.z = (bf16)f.z; o.w = (bf16)f.w;
    *(v4bf*)(dst + (size_t)i * 4) = o;
  }
}

// ---------------------------------------------------------------------------
// Tiled bf16 WMMA GEMM: C[M,1024] = A[M,1024] @ W[1024,1024]^T
//   block tile 128x128, K-tile 32; 8 waves (4 M-groups x 2 N-groups),
//   each wave computes 32x64 = 2x4 subtiles of 16x16.
//   Double-buffered LDS fed by async global->LDS copies: loads for tile k+1
//   overlap the WMMA work on tile k; one barrier per K-step.
// mode 0: write bf16 head-scattered [B,H,N,64], value *= scale  (QKV proj)
// mode 1: write f32 row-major [M,1024] + bias                    (out proj)
// ---------------------------------------------------------------------------
#define BM 128
#define BN 128
#define BK 32
#define LDSK 40   // 32 + 8 pad; 80-byte row stride keeps 16B alignment

__global__ __launch_bounds__(256) void gemm_bf16_wmma(
    const unsigned short* __restrict__ A_, const unsigned short* __restrict__ W_,
    unsigned short* __restrict__ outB_, float* __restrict__ outF,
    const float* __restrict__ bias, float scale, int mode) {
  const bf16* A = (const bf16*)A_;
  const bf16* W = (const bf16*)W_;
  bf16* outB = (bf16*)outB_;

  __shared__ bf16 sA[2][BM][LDSK];
  __shared__ bf16 sB[2][BN][LDSK];

  const int tid  = threadIdx.x;
  const int lane = tid & 31, wid = tid >> 5;
  const int half = lane >> 4, r = lane & 15;
  const int wm = wid & 3, wn = wid >> 2;
  const int bm = blockIdx.y * BM, bn = blockIdx.x * BN;
  const int K = D_MODEL;
  const int NK = K / BK;

  const v8f vzero = {};
  v8f acc[2][4];
  for (int i = 0; i < 2; ++i)
    for (int j = 0; j < 4; ++j) acc[i][j] = vzero;

  const int lr = tid >> 2;         // 0..63
  const int lc = (tid & 3) * 8;    // 0,8,16,24

  auto issue_tile = [&](int buf, int k0) {
    copy_b128_to_lds(&sA[buf][lr     ][lc], &A[(size_t)(bm + lr     ) * K + k0 + lc]);
    copy_b128_to_lds(&sA[buf][lr + 64][lc], &A[(size_t)(bm + lr + 64) * K + k0 + lc]);
    copy_b128_to_lds(&sB[buf][lr     ][lc], &W[(size_t)(bn + lr     ) * K + k0 + lc]);
    copy_b128_to_lds(&sB[buf][lr + 64][lc], &W[(size_t)(bn + lr + 64) * K + k0 + lc]);
  };

  issue_tile(0, 0);
  for (int kt = 0; kt < NK; ++kt) {
    wait_async0();      // my async writes for buf (kt&1) are in LDS
    __syncthreads();    // everyone's writes visible; previous compute finished
    if (kt + 1 < NK) issue_tile((kt + 1) & 1, (kt + 1) * BK);  // overlaps WMMA below

    const int cb = kt & 1;
    v16bf af[2], bfv[4];
    for (int i = 0; i < 2; ++i) af[i]  = frag_a(&sA[cb][wm * 32 + i * 16 + r][0], half);
    for (int j = 0; j < 4; ++j) bfv[j] = frag_b(&sB[cb][wn * 64 + j * 16 + r][0], half);
    for (int i = 0; i < 2; ++i)
      for (int j = 0; j < 4; ++j)
        acc[i][j] = wmma_bf16(af[i], bfv[j], acc[i][j]);
  }

  // epilogue (mode branch hoisted out of the store loops)
  const int rowb0 = bm + wm * 32 + half * 8;
  const int colb0 = bn + wn * 64 + r;
  if (mode == 0) {
    for (int i = 0; i < 2; ++i)
      for (int j = 0; j < 4; ++j) {
        const int rowb = rowb0 + i * 16;
        const int col  = colb0 + j * 16;
        const int h = col >> 6, d = col & (DHEAD - 1);
        for (int v = 0; v < 8; ++v) {
          const int rr = rowb + v;
          const int b = rr >> 11, n = rr & (NSEQ - 1);
          outB[(((size_t)(b * NHEADS + h) * NSEQ + n) << 6) + d] =
              (bf16)(acc[i][j][v] * scale);
        }
      }
  } else {
    for (int i = 0; i < 2; ++i)
      for (int j = 0; j < 4; ++j) {
        const int rowb = rowb0 + i * 16;
        const int col  = colb0 + j * 16;
        const float bv = bias[col];
        for (int v = 0; v < 8; ++v)
          outF[(size_t)(rowb + v) * D_MODEL + col] = acc[i][j][v] * scale + bv;
      }
  }
}

// ---------------------------------------------------------------------------
// Flash attention with WMMA. Grid: (N/64 query tiles, B*H). Block: 128 thr.
// Each of 4 waves owns 16 query rows: S=Q@K^T (8 wmma), masked online
// softmax with in-wave shfl_xor reductions, P staged via LDS, O+=P@V (8 wmma).
// Q and K tiles arrive via async global->LDS; V is transposed through VGPRs.
// qh/kh/vh layout: [B*H, N, 64] bf16.  Output o: [B, N, H*64] bf16.
// ---------------------------------------------------------------------------
__global__ __launch_bounds__(128) void attn_wmma(
    const unsigned short* __restrict__ Q_, const unsigned short* __restrict__ K_,
    const unsigned short* __restrict__ V_, const unsigned char* __restrict__ mask,
    unsigned short* __restrict__ O_) {
  const bf16* Q = (const bf16*)Q_;
  const bf16* Kh = (const bf16*)K_;
  const bf16* V = (const bf16*)V_;
  bf16* O = (bf16*)O_;

  __shared__ bf16 sQ[64][72];
  __shared__ bf16 sK[64][72];
  __shared__ bf16 sV[64][72];      // transposed: sV[d][key]
  __shared__ bf16 sP[4][16][72];   // per-wave P staging

  const int tid = threadIdx.x, wid = tid >> 5, lane = tid & 31;
  const int half = lane >> 4, r = lane & 15;
  const int qbase = blockIdx.x * 64;
  const int bh = blockIdx.y;
  const int b = bh >> 4, h = bh & (NHEADS - 1);
  const size_t headOff = (size_t)bh * NSEQ * DHEAD;

  const int lrow = tid >> 1;            // 0..63
  const int lcol = (tid & 1) * 32;      // 0 or 32

  { // Q tile (64x64): async, consumed after first-iteration wait+barrier
    const bf16* src = Q + headOff + (size_t)(qbase + lrow) * DHEAD + lcol;
    for (int j = 0; j < 32; j += 8) copy_b128_to_lds(&sQ[lrow][lcol + j], src + j);
  }

  float mstate[8], lstate[8];
  for (int v = 0; v < 8; ++v) { mstate[v] = -1e30f; lstate[v] = 0.f; }
  const v8f vzero = {};
  v8f oacc[4];
  for (int j = 0; j < 4; ++j) oacc[j] = vzero;

  for (int kt = 0; kt < NSEQ / 64; ++kt) {
    const int kb = kt * 64;
    __syncthreads();   // previous tile's compute done; safe to overwrite sK/sV
    { // K tile async; V tile transposed through VGPRs (overlaps async K)
      const bf16* ks = Kh + headOff + (size_t)(kb + lrow) * DHEAD + lcol;
      for (int j = 0; j < 32; j += 8) copy_b128_to_lds(&sK[lrow][lcol + j], ks + j);
      const bf16* vs = V + headOff + (size_t)(kb + lrow) * DHEAD + lcol;
      for (int j = 0; j < 32; ++j) sV[lcol + j][lrow] = vs[j];
    }
    wait_async0();
    __syncthreads();

    // ---- S = Q @ K^T : 16x64 per wave, f32 accum ----
    v16bf aq0 = frag_a(&sQ[wid * 16 + r][0], half);
    v16bf aq1 = frag_a(&sQ[wid * 16 + r][32], half);
    v8f s[4];
    for (int j = 0; j < 4; ++j) {
      v8f accv = vzero;
      accv = wmma_bf16(aq0, frag_b(&sK[j * 16 + r][0], half), accv);
      accv = wmma_bf16(aq1, frag_b(&sK[j * 16 + r][32], half), accv);
      s[j] = accv;
    }

    // ---- mask + per-row max ----
    float nmax[8];
    for (int v = 0; v < 8; ++v) nmax[v] = -3.0e38f;
    for (int j = 0; j < 4; ++j) {
      const int key = kb + j * 16 + r;
      const unsigned char* mp =
          mask + ((size_t)b * NSEQ + (qbase + wid * 16 + half * 8)) * NSEQ + key;
      for (int v = 0; v < 8; ++v) {
        float sv = s[j][v];
        if (mp[(size_t)v * NSEQ]) sv = -3.0e38f;
        s[j][v] = sv;
        nmax[v] = fmaxf(nmax[v], sv);
      }
    }
    for (int v = 0; v < 8; ++v) {      // reduce over the 16 lanes of this half
      float m = nmax[v];
      for (int sft = 1; sft < 16; sft <<= 1) m = fmaxf(m, __shfl_xor(m, sft, 32));
      nmax[v] = m;
    }

    // ---- online softmax update ----
    float fac[8], rsum[8];
    for (int v = 0; v < 8; ++v) {
      float nm = fmaxf(mstate[v], nmax[v]);
      fac[v] = __expf(mstate[v] - nm);
      mstate[v] = nm;
      rsum[v] = 0.f;
    }
    for (int j = 0; j < 4; ++j)
      for (int v = 0; v < 8; ++v) {
        float p = __expf(s[j][v] - mstate[v]);
        rsum[v] += p;
        sP[wid][half * 8 + v][j * 16 + r] = (bf16)p;   // stage P for A-frag reuse
      }
    for (int v = 0; v < 8; ++v) {
      float t = rsum[v];
      for (int sft = 1; sft < 16; sft <<= 1) t += __shfl_xor(t, sft, 32);
      lstate[v] = lstate[v] * fac[v] + t;
    }
    for (int j = 0; j < 4; ++j)
      for (int v = 0; v < 8; ++v) oacc[j][v] *= fac[v];

    // ---- O += P @ V ----  (same-wave LDS write->read; HW waits keep order)
    v16bf ap0 = frag_a(&sP[wid][r][0], half);
    v16bf ap1 = frag_a(&sP[wid][r][32], half);
    for (int j = 0; j < 4; ++j) {
      oacc[j] = wmma_bf16(ap0, frag_b(&sV[j * 16 + r][0], half), oacc[j]);
      oacc[j] = wmma_bf16(ap1, frag_b(&sV[j * 16 + r][32], half), oacc[j]);
    }
  }

  // ---- normalize and write o in [B, N, H*64] bf16 ----
  for (int j = 0; j < 4; ++j)
    for (int v = 0; v < 8; ++v) {
      int row = qbase + wid * 16 + half * 8 + v;
      int col = h * DHEAD + j * 16 + r;
      float val = oacc[j][v] / lstate[v];
      O[((size_t)(b * NSEQ + row) << 10) + col] = (bf16)val;
    }
}

// ---------------------------------------------------------------------------
// Launch. Workspace layout (needs 48 MiB):
//  [0,8M)  q bf16          [8M,10M) Wq bf16   [10M,12M) Wk   [12M,14M) Wv
//  [14M,16M) Wo bf16       [16M,24M) qh       [24M,32M) kh   [32M,40M) vh
//  [40M,48M) o bf16
// ---------------------------------------------------------------------------
extern "C" void kernel_launch(void* const* d_in, const int* in_sizes, int n_in,
                              void* d_out, int out_size, void* d_ws, size_t ws_size,
                              hipStream_t stream) {
  const float* q  = (const float*)d_in[0];
  const unsigned char* mask = (const unsigned char*)d_in[1];
  const float* Wq = (const float*)d_in[2];
  const float* Wk = (const float*)d_in[3];
  const float* Wv = (const float*)d_in[4];
  const float* Wo = (const float*)d_in[5];
  const float* bo = (const float*)d_in[6];
  float* out = (float*)d_out;

  char* ws = (char*)d_ws;
  const size_t MB = 1u << 20;
  unsigned short* qbf  = (unsigned short*)(ws + 0);
  unsigned short* wqbf = (unsigned short*)(ws + 8 * MB);
  unsigned short* wkbf = (unsigned short*)(ws + 10 * MB);
  unsigned short* wvbf = (unsigned short*)(ws + 12 * MB);
  unsigned short* wobf = (unsigned short*)(ws + 14 * MB);
  unsigned short* qh   = (unsigned short*)(ws + 16 * MB);
  unsigned short* kh   = (unsigned short*)(ws + 24 * MB);
  unsigned short* vh   = (unsigned short*)(ws + 32 * MB);
  unsigned short* ob   = (unsigned short*)(ws + 40 * MB);

  auto conv = [&](const float* s, unsigned short* d, int n) {
    int n4 = n / 4;
    cvt_f32_bf16<<<dim3((n4 + 255) / 256), 256, 0, stream>>>(s, d, n4);
  };
  conv(q,  qbf,  MROWS * D_MODEL);
  conv(Wq, wqbf, D_MODEL * D_MODEL);
  conv(Wk, wkbf, D_MODEL * D_MODEL);
  conv(Wv, wvbf, D_MODEL * D_MODEL);
  conv(Wo, wobf, D_MODEL * D_MODEL);

  const float SCALE = 0.125f;  // DIM_HEAD^-0.5 = 64^-0.5
  dim3 ggrid(D_MODEL / BN, MROWS / BM);  // (8, 32)
  gemm_bf16_wmma<<<ggrid, 256, 0, stream>>>(qbf, wqbf, qh, nullptr, nullptr, SCALE, 0);
  gemm_bf16_wmma<<<ggrid, 256, 0, stream>>>(qbf, wkbf, kh, nullptr, nullptr, 1.0f, 0);
  gemm_bf16_wmma<<<ggrid, 256, 0, stream>>>(qbf, wvbf, vh, nullptr, nullptr, 1.0f, 0);

  attn_wmma<<<dim3(NSEQ / 64, BATCH * NHEADS), 128, 0, stream>>>(qh, kh, vh, mask, ob);

  gemm_bf16_wmma<<<ggrid, 256, 0, stream>>>(ob, wobf, nullptr, out, bo, 1.0f, 1);
}
